// ALMSLayer_84911503441980
// MI455X (gfx1250) — matrix operand order repaired and checked by reading, not compiled, not run on hardware
//
#include <hip/hip_runtime.h>
#include <math.h>

#define BN 8192
#define DN 512
#define KNN 32

typedef __attribute__((ext_vector_type(16))) _Float16     v16h;
typedef __attribute__((ext_vector_type(8)))  float        v8f;
typedef __attribute__((ext_vector_type(4)))  unsigned int v4u;
typedef __attribute__((ext_vector_type(8)))  int          v8i;
typedef __attribute__((ext_vector_type(4)))  int          v4i;

#if defined(__has_builtin)
#if __has_builtin(__builtin_amdgcn_tensor_load_to_lds)
#define HAVE_TDM 1
#endif
#endif

// ---------------- WMMA tile helpers (wave32, 16x16x32 f16) ----------------
// A-tile (16x32): lane holds row L%16; h[0..7]=K[k+8*grp .. +8), h[8..15]=K[k+16+8*grp .. +8)
__device__ __forceinline__ void load_a_tile(const _Float16* arow, int k, int grp, v16h& a) {
  const _Float16* p = arow + k + (grp << 3);
#pragma unroll
  for (int t = 0; t < 8; ++t) { a[t] = p[t]; a[t + 8] = p[16 + t]; }
}
// B-tile (32x16) from row-major operand (NT gemm): lane holds col L%16; h[0..7]=K[k+8*grp .. +8), h[8..15]=K[k+16+8*grp .. +8)
__device__ __forceinline__ void load_b_tile(const _Float16* brow, int k, int grp, v16h& b) {
  const _Float16* p = brow + k + (grp << 3);
#pragma unroll
  for (int t = 0; t < 8; ++t) { b[t] = p[t]; b[t + 8] = p[16 + t]; }
}

// NT-gemm body: one wave computes a 16-row x (NT*16)-col strip, A loaded once per k-chunk.
template <int NT>
__device__ __forceinline__ void nt_gemm_body(const _Float16* __restrict__ A,
                                             const _Float16* __restrict__ B,
                                             int row0, int col0, int kdim, int lda, int ldb,
                                             int lane, v8f (&acc)[NT]) {
  const int grp = lane >> 4, lr = lane & 15;
  const _Float16* arow = A + (size_t)(row0 + lr) * lda;
  for (int k = 0; k < kdim; k += 32) {
    v16h a;
    load_a_tile(arow, k, grp, a);
#pragma unroll
    for (int c = 0; c < NT; ++c) {
      const _Float16* brow = B + (size_t)(col0 + c * 16 + lr) * ldb;
      v16h b;
      load_b_tile(brow, k, grp, b);
      acc[c] = __builtin_amdgcn_wmma_f32_16x16x32_f16(false, a, false, b, (short)0, acc[c],
                                                      false, false);
    }
  }
}

// ---------------- sim = x x^T (fp32 out), 4 column tiles per wave ----------------
__global__ void __launch_bounds__(32) sim_gemm_kernel(const _Float16* __restrict__ xh,
                                                      float* __restrict__ sim) {
  const int lane = threadIdx.x & 31, grp = lane >> 4, lr = lane & 15;
  const int tr = blockIdx.y, tc0 = blockIdx.x * 4;
  v8f acc[4] = {};
  nt_gemm_body<4>(xh, xh, tr * 16, tc0 * 16, DN, DN, DN, lane, acc);
#pragma unroll
  for (int c = 0; c < 4; ++c)
#pragma unroll
    for (int v = 0; v < 8; ++v)
      sim[(size_t)(tr * 16 + v + 8 * grp) * BN + (tc0 + c) * 16 + lr] = acc[c][v];
}

// ---------------- raw = (sim + 0.1*cos) / T, in place, 4 column tiles per wave ----------------
__global__ void __launch_bounds__(32) raw_gemm_kernel(const _Float16* __restrict__ fh,
                                                      const _Float16* __restrict__ gh,
                                                      const float* __restrict__ nf,
                                                      const float* __restrict__ ng,
                                                      float* __restrict__ sim) {
  const int lane = threadIdx.x & 31, grp = lane >> 4, lr = lane & 15;
  const int tr = blockIdx.y, tc0 = blockIdx.x * 4;
  v8f acc[4] = {};
  nt_gemm_body<4>(fh, gh, tr * 16, tc0 * 16, DN, DN, DN, lane, acc);
#pragma unroll
  for (int c = 0; c < 4; ++c)
#pragma unroll
    for (int v = 0; v < 8; ++v) {
      const int r = tr * 16 + v + 8 * grp, cc = (tc0 + c) * 16 + lr;
      const float cosv = acc[c][v] / fmaxf(nf[r] * ng[cc], 1e-8f);
      const size_t o = (size_t)r * BN + cc;
      sim[o] = (sim[o] + 0.1f * cosv) * 10.0f;   // 1/TEMPERATURE folded in
    }
}

// ---------------- enhanced = W @ F  (A fp32->f16 on the fly, B = F^T f16) ----------------
// 8 column tiles per wave: W (268MB, HBM-resident) is read only DN/128 = 4 times total.
__global__ void __launch_bounds__(32) out_gemm_kernel(const float* __restrict__ W,
                                                      const _Float16* __restrict__ fT,
                                                      float* __restrict__ out) {
  const int lane = threadIdx.x & 31, grp = lane >> 4, lr = lane & 15;
  const int tr = blockIdx.y;
  const int tc0 = blockIdx.x * 8;             // grid.x = DN/128 = 4
  const float* arow = W + (size_t)(tr * 16 + lr) * BN;
  v8f acc[8] = {};
  for (int k = 0; k < BN; k += 32) {
    v16h a;
    {
      const float* p = arow + k + (grp << 3);
      __builtin_prefetch(p + 512, 0, 1);      // next k-chunks of W -> global_prefetch_b8
#pragma unroll
      for (int t = 0; t < 8; ++t) { a[t] = (_Float16)p[t]; a[t + 8] = (_Float16)p[16 + t]; }
    }
#pragma unroll
    for (int c = 0; c < 8; ++c) {
      const _Float16* brow = fT + (size_t)((tc0 + c) * 16 + lr) * BN;
      v16h b;
      load_b_tile(brow, k, grp, b);
      acc[c] = __builtin_amdgcn_wmma_f32_16x16x32_f16(false, a, false, b, (short)0, acc[c],
                                                      false, false);
    }
  }
#pragma unroll
  for (int c = 0; c < 8; ++c)
#pragma unroll
    for (int v = 0; v < 8; ++v)
      out[(size_t)(tr * 16 + v + 8 * grp) * DN + (tc0 + c) * 16 + lr] = acc[c][v];
}

// ---------------- row norms ----------------
__global__ void __launch_bounds__(256) rownorm_kernel(const float* __restrict__ X,
                                                      float* __restrict__ nrm) {
  __shared__ float red[256];
  const int r = blockIdx.x, tid = threadIdx.x;
  const float* row = X + (size_t)r * DN;
  float s = 0.f;
  for (int i = tid; i < DN; i += 256) { float v = row[i]; s += v * v; }
  red[tid] = s; __syncthreads();
  for (int off = 128; off > 0; off >>= 1) {
    if (tid < off) red[tid] += red[tid + off];
    __syncthreads();
  }
  if (tid == 0) nrm[r] = sqrtf(red[0]);
}

// ---------------- prep: f16 copies + normalized rows + transpose ----------------
__global__ void __launch_bounds__(256) prep_kernel(const float* __restrict__ F,
                                                   const float* __restrict__ nf,
                                                   _Float16* __restrict__ xh,
                                                   _Float16* __restrict__ fh,
                                                   _Float16* __restrict__ fT) {
  const size_t idx = (size_t)blockIdx.x * 256 + threadIdx.x;
  if (idx >= (size_t)BN * DN) return;
  const int r = (int)(idx / DN), c = (int)(idx % DN);
  const float v = F[idx];
  fh[idx] = (_Float16)v;
  xh[idx] = (_Float16)(v / fmaxf(nf[r], 1e-12f));
  fT[(size_t)c * BN + r] = (_Float16)v;
}

__global__ void __launch_bounds__(256) cvt16_kernel(const float* __restrict__ src,
                                                    _Float16* __restrict__ dst, size_t n) {
  const size_t i = (size_t)blockIdx.x * 256 + threadIdx.x;
  if (i < n) dst[i] = (_Float16)src[i];
}

// ---------------- top-K per row (self excluded) via TDM-staged LDS argmax ----------------
__global__ void __launch_bounds__(256) topk_kernel(const float* __restrict__ sim,
                                                   int* __restrict__ topk) {
  __shared__ float srow[BN];
  __shared__ float rv[256];
  __shared__ int   ri[256];
  const int r = blockIdx.x, tid = threadIdx.x;
  const float* row = sim + (size_t)r * BN;

#if defined(HAVE_TDM)
  // Stage the 32KB sim row into LDS with the Tensor Data Mover (2D D#, one row of
  // 8192 fp32). Issued from wave 0 only; TDM ignores EXEC. TENSORcnt guards reuse.
  if (tid == 0) {
    const unsigned long long ga = (unsigned long long)(uintptr_t)row;
    const unsigned int ldsOff = (unsigned int)(uintptr_t)(void*)srow;  // low 32b = LDS offset
    v4u g0 = { 1u,                                        // count=1, user descriptor
               ldsOff,                                     // lds_addr
               (unsigned int)(ga & 0xffffffffu),           // global_addr[31:0]
               (unsigned int)(((ga >> 32) & 0x01ffffffu) | (2u << 30)) }; // addr[56:32]|type=2
    v8i g1 = { (int)(2u << 16),                 // workgroup_mask=0, data_size=2 (4B)
               (int)((BN & 0xffffu) << 16),     // tensor_dim0[15:0]
               (int)((BN >> 16) | (1u << 16)),  // tensor_dim0[31:16] | tensor_dim1[15:0]=1
               (int)((BN & 0xffffu) << 16),     // tensor_dim1[31:16]=0 | tile_dim0=8192
               1,                               // tile_dim1=1, tile_dim2=0
               (int)BN,                         // tensor_dim0_stride[31:0]
               (int)((BN & 0xffffu) << 16),     // stride0[47:32]=0 | stride1[15:0]
               0 };                             // stride1[47:16]=0
    v4i z4 = {0, 0, 0, 0};
#if __clang_major__ >= 23
    v8i z8 = {0, 0, 0, 0, 0, 0, 0, 0};
    __builtin_amdgcn_tensor_load_to_lds(g0, g1, z4, z4, z8, 0);
#else
    __builtin_amdgcn_tensor_load_to_lds(g0, g1, z4, z4, 0);
#endif
#if __has_builtin(__builtin_amdgcn_s_wait_tensorcnt)
    __builtin_amdgcn_s_wait_tensorcnt(0);
#else
    asm volatile("s_wait_tensorcnt 0x0" ::: "memory");
#endif
  }
  __syncthreads();
  if (tid == 0) srow[r] = -INFINITY;            // exclude self
  __syncthreads();
#else
  for (int i = tid; i < BN; i += 256) srow[i] = (i == r) ? -INFINITY : row[i];
  __syncthreads();
#endif

  for (int s = 0; s < KNN; ++s) {
    float bv = -INFINITY; int bi = 0;
    for (int i = tid; i < BN; i += 256) {
      const float v = srow[i];
      if (v > bv) { bv = v; bi = i; }
    }
    rv[tid] = bv; ri[tid] = bi;
    __syncthreads();
    for (int off = 128; off > 0; off >>= 1) {
      if (tid < off) {
        if (rv[tid + off] > rv[tid] ||
            (rv[tid + off] == rv[tid] && ri[tid + off] < ri[tid])) {
          rv[tid] = rv[tid + off]; ri[tid] = ri[tid + off];
        }
      }
      __syncthreads();
    }
    if (tid == 0) { topk[r * KNN + s] = ri[0]; srow[ri[0]] = -INFINITY; }
    __syncthreads();
  }
}

// ---------------- degrees / normalization ----------------
__global__ void __launch_bounds__(256) deg_init_kernel(float* __restrict__ deg) {
  const int i = blockIdx.x * 256 + threadIdx.x;
  if (i < BN) deg[i] = (float)KNN;
}
__global__ void __launch_bounds__(256) deg_scatter_kernel(const int* __restrict__ topk,
                                                          float* __restrict__ deg) {
  const int i = blockIdx.x * 256 + threadIdx.x;
  if (i < BN * KNN) atomicAdd(&deg[topk[i]], 1.0f);
}
__global__ void __launch_bounds__(256) dinv_kernel(const float* __restrict__ deg,
                                                   float* __restrict__ dinv) {
  const int i = blockIdx.x * 256 + threadIdx.x;
  if (i < BN) dinv[i] = fminf(rsqrtf(deg[i]), 1e6f);
}

__global__ void __launch_bounds__(256) zero_kernel(float* __restrict__ p, size_t n) {
  const size_t i = (size_t)blockIdx.x * 256 + threadIdx.x;
  if (i < n) p[i] = 0.f;
}

// ---------------- SpMM over symmetric kNN edges (one wave per directed pair) ----------------
__global__ void __launch_bounds__(256) spmm_kernel(const int* __restrict__ topk,
                                                   const float* __restrict__ dinv,
                                                   const float* __restrict__ src,
                                                   float* __restrict__ dst) {
  const int pair = blockIdx.x * 8 + (threadIdx.x >> 5);
  const int lane = threadIdx.x & 31;
  const int i = pair >> 5;        // KNN == 32
  const int kk = pair & 31;
  const int j = topk[i * KNN + kk];
  const float w = dinv[i] * dinv[j];
  const float* si = src + (size_t)i * DN;
  const float* sj = src + (size_t)j * DN;
  float* di = dst + (size_t)i * DN;
  float* dj = dst + (size_t)j * DN;
  for (int d = lane; d < DN; d += 32) {
    atomicAdd(di + d, w * sj[d]);   // edge (i <- j)
    atomicAdd(dj + d, w * si[d]);   // edge (j <- i)
  }
}

// ---------------- row softmax in place (input already scaled by 1/T) ----------------
__global__ void __launch_bounds__(256) softmax_kernel(float* __restrict__ W) {
  __shared__ float red[256];
  const int r = blockIdx.x, tid = threadIdx.x;
  float* row = W + (size_t)r * BN;
  float m = -INFINITY;
  for (int i = tid; i < BN; i += 256) m = fmaxf(m, row[i]);
  red[tid] = m; __syncthreads();
  for (int off = 128; off > 0; off >>= 1) {
    if (tid < off) red[tid] = fmaxf(red[tid], red[tid + off]);
    __syncthreads();
  }
  m = red[0]; __syncthreads();
  float s = 0.f;
  for (int i = tid; i < BN; i += 256) { const float e = __expf(row[i] - m); row[i] = e; s += e; }
  red[tid] = s; __syncthreads();
  for (int off = 128; off > 0; off >>= 1) {
    if (tid < off) red[tid] += red[tid + off];
    __syncthreads();
  }
  const float inv = 1.0f / red[0];
  for (int i = tid; i < BN; i += 256) row[i] *= inv;
}

// ---------------- orchestration ----------------
extern "C" void kernel_launch(void* const* d_in, const int* in_sizes, int n_in,
                              void* d_out, int out_size, void* d_ws, size_t ws_size,
                              hipStream_t stream) {
  (void)in_sizes; (void)n_in; (void)out_size; (void)ws_size;
  const float* feat = (const float*)d_in[0];
  float* out = (float*)d_out;

  char* w = (char*)d_ws;
  size_t off = 0;
  auto carve = [&](size_t bytes) -> void* {
    void* p = w + off;
    off += (bytes + 255) & ~(size_t)255;
    return p;
  };
  float*    sim  = (float*)   carve((size_t)BN * BN * sizeof(float));     // 256 MiB
  _Float16* xh   = (_Float16*)carve((size_t)BN * DN * sizeof(_Float16));
  _Float16* fh   = (_Float16*)carve((size_t)BN * DN * sizeof(_Float16));
  _Float16* fT   = (_Float16*)carve((size_t)BN * DN * sizeof(_Float16));
  _Float16* gh   = (_Float16*)carve((size_t)BN * DN * sizeof(_Float16));
  float*    g1   = (float*)   carve((size_t)BN * DN * sizeof(float));
  float*    geo  = (float*)   carve((size_t)BN * DN * sizeof(float));
  float*    nf   = (float*)   carve((size_t)BN * sizeof(float));
  float*    ng   = (float*)   carve((size_t)BN * sizeof(float));
  float*    deg  = (float*)   carve((size_t)BN * sizeof(float));
  float*    dinv = (float*)   carve((size_t)BN * sizeof(float));
  int*      topk = (int*)     carve((size_t)BN * KNN * sizeof(int));

  const int elems = BN * DN;                       // 4M
  const dim3 gemmGrid(BN / 64, BN / 16);           // 4 col tiles per wave

  rownorm_kernel<<<BN, 256, 0, stream>>>(feat, nf);
  prep_kernel<<<(elems + 255) / 256, 256, 0, stream>>>(feat, nf, xh, fh, fT);
  sim_gemm_kernel<<<gemmGrid, 32, 0, stream>>>(xh, sim);
  topk_kernel<<<BN, 256, 0, stream>>>(sim, topk);
  deg_init_kernel<<<(BN + 255) / 256, 256, 0, stream>>>(deg);
  deg_scatter_kernel<<<(BN * KNN + 255) / 256, 256, 0, stream>>>(topk, deg);
  dinv_kernel<<<(BN + 255) / 256, 256, 0, stream>>>(deg, dinv);
  zero_kernel<<<(elems + 255) / 256, 256, 0, stream>>>(g1, (size_t)elems);
  spmm_kernel<<<BN * KNN / 8, 256, 0, stream>>>(topk, dinv, feat, g1);
  zero_kernel<<<(elems + 255) / 256, 256, 0, stream>>>(geo, (size_t)elems);
  spmm_kernel<<<BN * KNN / 8, 256, 0, stream>>>(topk, dinv, g1, geo);
  rownorm_kernel<<<BN, 256, 0, stream>>>(geo, ng);
  cvt16_kernel<<<(elems + 255) / 256, 256, 0, stream>>>(geo, gh, (size_t)elems);
  raw_gemm_kernel<<<gemmGrid, 32, 0, stream>>>(fh, gh, nf, ng, sim);
  softmax_kernel<<<BN, 256, 0, stream>>>(sim);
  out_gemm_kernel<<<dim3(DN / 128, BN / 16), 32, 0, stream>>>(sim, fT, out);
}
